// GumbelTreeCell_31258771981069
// MI455X (gfx1250) — matrix operand order repaired and checked by reading, not compiled, not run on hardware
//
#include <hip/hip_runtime.h>
#include <math.h>

// Problem constants (reference: N,S,D = 16,32,256; CH = 1024)
#define N_  16
#define S_  32
#define D_  256
#define CH_ 1024

typedef __attribute__((ext_vector_type(16))) __bf16       v16bf;
typedef __attribute__((ext_vector_type(8)))  float        v8f;
typedef __attribute__((ext_vector_type(8)))  unsigned int v8u;

#define V8F_ZERO {0.f,0.f,0.f,0.f,0.f,0.f,0.f,0.f}

// ---------- small helpers ----------
__device__ __forceinline__ unsigned short f2bfu(float f) {
  unsigned u = __builtin_bit_cast(unsigned, f);
  unsigned r = u + 0x7FFFu + ((u >> 16) & 1u);   // round-to-nearest-even
  return (unsigned short)(r >> 16);
}
__device__ __forceinline__ float bf2f(unsigned short h) {
  unsigned u = ((unsigned)h) << 16;
  return __builtin_bit_cast(float, u);
}
__device__ __forceinline__ float wave_sum(float v) {
  #pragma unroll
  for (int off = 16; off > 0; off >>= 1) v += __shfl_xor(v, off, 32);
  return v;
}
__device__ __forceinline__ float sigmf(float x) { return 1.f / (1.f + __expf(-x)); }
__device__ __forceinline__ float geluf(float x) {
  return 0.5f * x * (1.f + erff(x * 0.70710678118654752f));
}

// ---------- WMMA fragment loaders (CDNA5 16-bit layouts, wave32) ----------
// A (16x32 bf16): lane L -> M = L&15; VGPR r holds K = 16*(r>>2) + 8*(L>>4) + 2*(r&3) + {0,1}
__device__ __forceinline__ v16bf load_a_bf(const unsigned short* base, int stride,
                                           int m0, int kb, int lane) {
  const unsigned short* rowp = base + (m0 + (lane & 15)) * stride + kb;
  int hk = (lane >> 4) << 3;
  v8u au;
  #pragma unroll
  for (int r = 0; r < 8; ++r) {
    int k = ((r & 4) << 2) + hk + ((r & 3) << 1);
    au[r] = *(const unsigned*)(const void*)(rowp + k);   // 2 packed bf16
  }
  return __builtin_bit_cast(v16bf, au);
}
// B (32x16 bf16): lane L -> N = L&15; VGPR r holds K = 16*(r>>2) + 8*(L>>4) + 2*(r&3) + {0,1}
__device__ __forceinline__ v16bf load_b_glob(const unsigned short* base, int stride,
                                             int kb, int n0, int lane) {
  const unsigned short* p = base + (unsigned)kb * stride + n0 + (lane & 15);
  int hk = (lane >> 4) << 3;
  v8u bu;
  #pragma unroll
  for (int r = 0; r < 8; ++r) {
    int k = ((r & 4) << 2) + hk + ((r & 3) << 1);
    bu[r] = (unsigned)p[(unsigned)k * stride] | ((unsigned)p[(unsigned)(k + 1) * stride] << 16);
  }
  return __builtin_bit_cast(v16bf, bu);
}
__device__ __forceinline__ v8f wmma_bf16(v16bf a, v16bf b, v8f c) {
  return __builtin_amdgcn_wmma_f32_16x16x32_bf16(false, a, false, b, (short)0, c,
                                                 false, false);
}

// ---------- kernel 0: one-time weight conversion f32 -> bf16 ----------
__global__ void prep_kernel(const float* __restrict__ ww, const float* __restrict__ wc1,
                            const float* __restrict__ wc2,
                            unsigned short* o_ww, unsigned short* o_c1, unsigned short* o_c2) {
  const int n0 = D_ * D_;          // 65536
  const int n1 = 2 * D_ * CH_;     // 524288
  const int n2 = CH_ * 4 * D_;     // 1048576
  int i = blockIdx.x * blockDim.x + threadIdx.x;
  for (; i < n0 + n1 + n2; i += gridDim.x * blockDim.x) {
    if (i < n0)            o_ww[i]           = f2bfu(ww[i]);
    else if (i < n0 + n1)  o_c1[i - n0]      = f2bfu(wc1[i - n0]);
    else                   o_c2[i - n0 - n1] = f2bfu(wc2[i - n0 - n1]);
  }
}

// ---------- kernel 1: state0 = LN(x @ W_word + b_word), one block per batch ----------
__global__ __launch_bounds__(256) void embed_kernel(
    const float* __restrict__ x, const unsigned short* __restrict__ Wwb,
    const float* __restrict__ bw, const float* __restrict__ lnw,
    const float* __restrict__ lnb, float* __restrict__ stateOut) {
  __shared__ __align__(32) unsigned short s_xa[32 * D_];   // 16 KB bf16 copy of x[n]
  __shared__ float s_h[32 * D_];                           // 32 KB pre-LN activations
  const int n = blockIdx.x, tid = threadIdx.x;
  const int lane = tid & 31, wave = tid >> 5;

  for (int e = tid; e < 32 * D_; e += 256) s_xa[e] = f2bfu(x[n * (S_ * D_) + e]);
  __syncthreads();

  // 2(M) x 2(N) register-blocked tiles: wave w owns cols [w*32, w*32+32)
  {
    v8f acc[4] = {V8F_ZERO, V8F_ZERO, V8F_ZERO, V8F_ZERO};
    int nbase = wave << 5;
    #pragma unroll 2
    for (int kb = 0; kb < D_; kb += 32) {
      v16bf a0 = load_a_bf(s_xa, D_, 0, kb, lane);
      v16bf a1 = load_a_bf(s_xa, D_, 16, kb, lane);
      #pragma unroll
      for (int j = 0; j < 2; ++j) {
        v16bf b = load_b_glob(Wwb, D_, kb, nbase + (j << 4), lane);
        acc[j]     = wmma_bf16(a0, b, acc[j]);
        acc[2 + j] = wmma_bf16(a1, b, acc[2 + j]);
      }
    }
    int rb0 = (lane >> 4) << 3;
    #pragma unroll
    for (int j = 0; j < 2; ++j) {
      int col = nbase + (j << 4) + (lane & 15);
      float bias = bw[col];
      #pragma unroll
      for (int r = 0; r < 8; ++r) {
        s_h[(rb0 + r) * D_ + col]      = acc[j][r]     + bias;
        s_h[(16 + rb0 + r) * D_ + col] = acc[2 + j][r] + bias;
      }
    }
  }
  __syncthreads();

  for (int m = wave; m < 32; m += 8) {            // LN per row, one wave per row
    float hv[8], s1 = 0.f, s2 = 0.f;
    #pragma unroll
    for (int jj = 0; jj < 8; ++jj) {
      int d = lane + (jj << 5);
      float h = s_h[m * D_ + d];
      hv[jj] = h; s1 += h; s2 += h * h;
    }
    s1 = wave_sum(s1); s2 = wave_sum(s2);
    float mean = s1 * (1.f / 256.f);
    float var  = s2 * (1.f / 256.f) - mean * mean;
    float inv  = rsqrtf(var + 1e-5f);
    #pragma unroll
    for (int jj = 0; jj < 8; ++jj) {
      int d = lane + (jj << 5);
      stateOut[n * (S_ * D_) + m * D_ + d] = (hv[jj] - mean) * inv * lnw[d] + lnb[d];
    }
  }
}

// ---------- kernel 2: one full tree-merge iteration, one block per batch ----------
__global__ __launch_bounds__(256) void merge_kernel(
    const float* __restrict__ stateIn, float* __restrict__ stateOut,
    const unsigned short* __restrict__ Wc1, const float* __restrict__ bc1,
    const unsigned short* __restrict__ Wc2, const float* __restrict__ bc2,
    const float* __restrict__ ln2w, const float* __restrict__ ln2b,
    const float* __restrict__ Wdec, const float* __restrict__ bdec,
    const float* __restrict__ mask, int it, int M, int last) {
  // 160 KB LDS arena, phase-lifetime overlays:
  //   [0      , 32768 )  f32  state[32][256]     (live whole kernel)
  //   [32768  , 98304 )  bf16 inter[32][1024]    (phase B/C)   overlaid in D/E by
  //   [32768  , 65536 )  f32  nh[32][256]
  //   [98304  ,163840 )  bf16 cont[32][1024]     (phase C/D)   overlaid in A2/B by
  //   [98304  ,131072 )  bf16 cat[32][512]       and in E by logit/maskv/k scalars
  __shared__ __align__(32) char arena[163840];
  float*          s_state = (float*)arena;
  unsigned short* s_inter = (unsigned short*)(arena + 32768);
  float*          s_nh    = (float*)(arena + 32768);
  unsigned short* s_cont  = (unsigned short*)(arena + 98304);
  unsigned short* s_cat   = (unsigned short*)(arena + 98304);
  float*          s_logit = (float*)(arena + 98304);
  float*          s_maskv = (float*)(arena + 98304 + 128);
  int*            s_k     = (int*)(arena + 98304 + 256);

  const int n = blockIdx.x, tid = threadIdx.x;
  const int lane = tid & 31, wave = tid >> 5;
  const int useHi = (M > 16);

  // Phase A: stage (M+1) state rows
  for (int e = tid; e < (M + 1) * D_; e += 256)
    s_state[e] = stateIn[n * (S_ * D_) + e];
  __syncthreads();

  // Phase A2: materialize cat(l,r) once as bf16 (zero-padded rows >= M)
  for (int e = tid; e < 32 * 2 * D_; e += 256) {
    int m = e >> 9, c = e & 511;
    float v = (m < M) ? s_state[(m + (c >> 8)) * D_ + (c & 255)] : 0.f;
    s_cat[e] = f2bfu(v);
  }
  __syncthreads();

  // Phase B: inter = gelu(cat @ Wc1 + bc1)   [M x 1024]
  // Register blocking: 2(M) x 4(N) tiles per wave, 2 groups -> wave owns 128 cols.
  for (int g = 0; g < 2; ++g) {
    int nbase = (wave << 7) + (g << 6);
    v8f acc[8] = {V8F_ZERO,V8F_ZERO,V8F_ZERO,V8F_ZERO,V8F_ZERO,V8F_ZERO,V8F_ZERO,V8F_ZERO};
    #pragma unroll 2
    for (int kb = 0; kb < 2 * D_; kb += 32) {
      v16bf a0 = load_a_bf(s_cat, 2 * D_, 0, kb, lane);
      v16bf a1 = a0;
      if (useHi) a1 = load_a_bf(s_cat, 2 * D_, 16, kb, lane);
      #pragma unroll
      for (int j = 0; j < 4; ++j) {
        v16bf b = load_b_glob(Wc1, CH_, kb, nbase + (j << 4), lane);
        acc[j] = wmma_bf16(a0, b, acc[j]);
        if (useHi) acc[4 + j] = wmma_bf16(a1, b, acc[4 + j]);
      }
    }
    int rb0 = (lane >> 4) << 3;
    #pragma unroll
    for (int j = 0; j < 4; ++j) {
      int col = nbase + (j << 4) + (lane & 15);
      float bias = bc1[col];
      #pragma unroll
      for (int r = 0; r < 8; ++r)
        s_inter[(rb0 + r) * CH_ + col] = f2bfu(geluf(acc[j][r] + bias));
      if (useHi) {
        #pragma unroll
        for (int r = 0; r < 8; ++r)
          s_inter[(16 + rb0 + r) * CH_ + col] = f2bfu(geluf(acc[4 + j][r] + bias));
      }
    }
  }
  __syncthreads();

  // Phase C: cont = inter @ Wc2 + bc2   [M x 1024]   (overwrites cat region)
  for (int g = 0; g < 2; ++g) {
    int nbase = (wave << 7) + (g << 6);
    v8f acc[8] = {V8F_ZERO,V8F_ZERO,V8F_ZERO,V8F_ZERO,V8F_ZERO,V8F_ZERO,V8F_ZERO,V8F_ZERO};
    #pragma unroll 2
    for (int kb = 0; kb < CH_; kb += 32) {
      v16bf a0 = load_a_bf(s_inter, CH_, 0, kb, lane);
      v16bf a1 = a0;
      if (useHi) a1 = load_a_bf(s_inter, CH_, 16, kb, lane);
      #pragma unroll
      for (int j = 0; j < 4; ++j) {
        v16bf b = load_b_glob(Wc2, CH_, kb, nbase + (j << 4), lane);
        acc[j] = wmma_bf16(a0, b, acc[j]);
        if (useHi) acc[4 + j] = wmma_bf16(a1, b, acc[4 + j]);
      }
    }
    int rb0 = (lane >> 4) << 3;
    #pragma unroll
    for (int j = 0; j < 4; ++j) {
      int col = nbase + (j << 4) + (lane & 15);
      float bias = bc2[col];
      #pragma unroll
      for (int r = 0; r < 8; ++r)
        s_cont[(rb0 + r) * CH_ + col] = f2bfu(acc[j][r] + bias);
      if (useHi) {
        #pragma unroll
        for (int r = 0; r < 8; ++r)
          s_cont[(16 + rb0 + r) * CH_ + col] = f2bfu(acc[4 + j][r] + bias);
      }
    }
  }
  __syncthreads();

  // Phase D: nh = LN(sig(g0)*l + sig(g1)*r + sig(g2)*parent)  (unmasked)
  for (int m = wave; m < M; m += 8) {
    float hv[8], s1 = 0.f, s2 = 0.f;
    #pragma unroll
    for (int jj = 0; jj < 8; ++jj) {
      int d = lane + (jj << 5);
      float c0 = bf2f(s_cont[m * CH_ + d]);
      float c1 = bf2f(s_cont[m * CH_ + 256 + d]);
      float c2 = bf2f(s_cont[m * CH_ + 512 + d]);
      float pp = bf2f(s_cont[m * CH_ + 768 + d]);
      float lv = s_state[m * D_ + d];
      float rv = s_state[(m + 1) * D_ + d];
      float h = sigmf(c0) * lv + sigmf(c1) * rv + sigmf(c2) * pp;
      hv[jj] = h; s1 += h; s2 += h * h;
    }
    s1 = wave_sum(s1); s2 = wave_sum(s2);
    float mean = s1 * (1.f / 256.f);
    float var  = s2 * (1.f / 256.f) - mean * mean;
    float inv  = rsqrtf(var + 1e-5f);
    #pragma unroll
    for (int jj = 0; jj < 8; ++jj) {
      int d = lane + (jj << 5);
      s_nh[m * D_ + d] = (hv[jj] - mean) * inv * ln2w[d] + ln2b[d];
    }
  }
  __syncthreads();

  float done = mask[n * S_ + it + 1];

  if (!last) {
    if (tid < M) s_maskv[tid] = mask[n * S_ + it + 1 + tid];
    __syncthreads();
    // decode logits: win = [nh[m-2],nh[m-1],nh[m],nh[m+1],nh[m+2]] . W_dec
    for (int m = wave; m < M; m += 8) {
      float acc = 0.f;
      #pragma unroll
      for (int c = 0; c < 5; ++c) {
        int src = m + c - 2;
        if (src >= 0 && src < M) {
          float mv = s_maskv[src];
          #pragma unroll
          for (int jj = 0; jj < 8; ++jj) {
            int d = lane + (jj << 5);
            acc += s_nh[src * D_ + d] * mv * Wdec[c * D_ + d];
          }
        }
      }
      acc = wave_sum(acc);
      if (lane == 0) s_logit[m] = (acc + bdec[0]) * 0.027950849718747371f; // 1/sqrt(1280)
    }
    __syncthreads();
    // hard selection: forward value of sel == one-hot(argmax of masked softmax)
    if (tid == 0) {
      int best = 0; float bv = -3.0e38f;
      for (int m = 0; m < M; ++m) {
        float key = (s_maskv[m] > 0.f) ? s_logit[m] : -3.0e38f;
        if (key > bv) { bv = key; best = m; }
      }
      *s_k = best;
    }
    __syncthreads();
    int k = *s_k;
    for (int e = tid; e < M * D_; e += 256) {
      int m = e >> 8, d = e & 255;
      float outv;
      if (m == k)      outv = s_nh[m * D_ + d] * s_maskv[m];   // merged node
      else if (m < k)  outv = s_state[m * D_ + d];             // keep left
      else             outv = s_state[(m + 1) * D_ + d];       // shift right
      stateOut[n * (S_ * D_) + e] = done * outv + (1.f - done) * s_state[e];
    }
  } else {
    // final merge (M == 1): no selection
    for (int e = tid; e < D_; e += 256)
      stateOut[n * (S_ * D_) + e] = done * s_nh[e] + (1.f - done) * s_state[e];
  }
}

// ---------- kernel 3: pack outputs (x, global_state) ----------
__global__ void finalize_kernel(const float* __restrict__ x,
                                const float* __restrict__ stateF,
                                float* __restrict__ out) {
  const int totX = N_ * S_ * D_;
  const int tot  = totX + N_ * D_;
  int i = blockIdx.x * blockDim.x + threadIdx.x;
  for (; i < tot; i += gridDim.x * blockDim.x) {
    if (i < totX) out[i] = x[i];
    else {
      int e = i - totX;
      out[i] = stateF[(e >> 8) * (S_ * D_) + (e & 255)];
    }
  }
}

extern "C" void kernel_launch(void* const* d_in, const int* in_sizes, int n_in,
                              void* d_out, int out_size, void* d_ws, size_t ws_size,
                              hipStream_t stream) {
  (void)in_sizes; (void)n_in; (void)out_size; (void)ws_size;
  const float* x     = (const float*)d_in[0];
  const float* mask  = (const float*)d_in[1];
  const float* Wword = (const float*)d_in[2];
  const float* bword = (const float*)d_in[3];
  const float* lnw   = (const float*)d_in[4];
  const float* lnb   = (const float*)d_in[5];
  const float* Wc1   = (const float*)d_in[6];
  const float* bc1   = (const float*)d_in[7];
  const float* Wc2   = (const float*)d_in[8];
  const float* bc2   = (const float*)d_in[9];
  const float* ln2w  = (const float*)d_in[10];
  const float* ln2b  = (const float*)d_in[11];
  const float* Wdec  = (const float*)d_in[12];
  const float* bdec  = (const float*)d_in[13];

  // Workspace layout (≈4.3 MB): bf16 weights + ping-pong f32 state buffers.
  char* ws = (char*)d_ws;
  unsigned short* wword_bf = (unsigned short*)(ws + 0);        // 128 KB
  unsigned short* wc1_bf   = (unsigned short*)(ws + 131072);   // 1 MB
  unsigned short* wc2_bf   = (unsigned short*)(ws + 1179648);  // 2 MB
  float* stateA = (float*)(ws + 3276800);                      // 512 KB
  float* stateB = (float*)(ws + 3801088);                      // 512 KB

  prep_kernel<<<640, 256, 0, stream>>>(Wword, Wc1, Wc2, wword_bf, wc1_bf, wc2_bf);
  embed_kernel<<<N_, 256, 0, stream>>>(x, wword_bf, bword, lnw, lnb, stateA);
  for (int i = 0; i < S_ - 1; ++i) {
    const float* sin  = (i & 1) ? stateB : stateA;
    float*       sout = (i & 1) ? stateA : stateB;
    merge_kernel<<<N_, 256, 0, stream>>>(sin, sout, wc1_bf, bc1, wc2_bf, bc2,
                                         ln2w, ln2b, Wdec, bdec, mask,
                                         i, S_ - 1 - i, (i == S_ - 2) ? 1 : 0);
  }
  finalize_kernel<<<528, 256, 0, stream>>>(x, stateB, (float*)d_out);  // 31 iters -> stateB
}
